// TransformerBlock_38182259261893
// MI455X (gfx1250) — matrix operand order repaired and checked by reading, not compiled, not run on hardware
//
#include <hip/hip_runtime.h>
#include <cstdint>
#include <cstddef>

// ---------------------------------------------------------------------------
// Types for CDNA5 WMMA (wave32). Trivially-copyable ext-vectors (union-safe).
// ---------------------------------------------------------------------------
typedef __attribute__((ext_vector_type(16))) __bf16 v16bf;
typedef __attribute__((ext_vector_type(8)))  float  v8f;
typedef __attribute__((ext_vector_type(4)))  unsigned int v4u;
typedef __attribute__((ext_vector_type(8)))  int    v8i;
typedef __attribute__((ext_vector_type(4)))  int    v4i;

union FragBF { v4u u[2]; v16bf v; };

#define AS1 __attribute__((address_space(1)))
#define AS3 __attribute__((address_space(3)))

// CDNA5 async / TDM availability (device pass only; host pass parses fallback)
#if __has_builtin(__builtin_amdgcn_global_load_async_to_lds_b128)
#define HAVE_ASYNC_LDS 1
#endif
#if __has_builtin(__builtin_amdgcn_tensor_load_to_lds)
#define HAVE_TDM 1
#endif

static constexpr int B_   = 2;
static constexpr int S_   = 2048;
static constexpr int H_   = 1024;
static constexpr int NH_  = 16;
static constexpr int HD_  = 64;
static constexpr int FFN_ = 4096;
static constexpr int M_   = B_ * S_;   // 4096 activation rows

// ---------------------------------------------------------------------------
// Wait helpers for async / tensor staging
// ---------------------------------------------------------------------------
__device__ __forceinline__ void wait_staging() {
#if defined(HAVE_ASYNC_LDS)
#if __has_builtin(__builtin_amdgcn_s_wait_asynccnt)
  __builtin_amdgcn_s_wait_asynccnt(0);
#else
  asm volatile("s_wait_asynccnt 0x0" ::: "memory");
#endif
#endif
#if defined(HAVE_TDM)
  __builtin_amdgcn_s_wait_tensorcnt(0);
#endif
}

// ---------------------------------------------------------------------------
// fp32 -> bf16 cast + transpose: W[K][N] -> Wt[N][K]
// ---------------------------------------------------------------------------
__global__ void cast_transpose_bf16_kernel(const float* __restrict__ src,
                                           __bf16* __restrict__ dst,
                                           int K, int N) {
  int idx = blockIdx.x * blockDim.x + threadIdx.x;  // K*N threads
  int k = idx / N, n = idx - k * N;
  dst[(size_t)n * K + k] = (__bf16)src[idx];
}

// ---------------------------------------------------------------------------
// RMSNorm: one 256-thread block per row of H_=1024; writes bf16.
// ---------------------------------------------------------------------------
__global__ __launch_bounds__(256)
void rmsnorm_bf16_kernel(const float* __restrict__ x,
                         const float* __restrict__ g,
                         __bf16* __restrict__ out) {
  __shared__ float red[8];
  const int row = blockIdx.x;
  const int tid = threadIdx.x;
  const float* xr = x + (size_t)row * H_;
  float4 f = *(const float4*)(xr + tid * 4);
  float ss = f.x * f.x + f.y * f.y + f.z * f.z + f.w * f.w;
  #pragma unroll
  for (int m = 1; m < 32; m <<= 1) ss += __shfl_xor(ss, m, 32);
  if ((tid & 31) == 0) red[tid >> 5] = ss;
  __syncthreads();
  float tot = red[0] + red[1] + red[2] + red[3] +
              red[4] + red[5] + red[6] + red[7];
  float inv = rsqrtf(tot / (float)H_ + 1e-6f);
  float4 gv = *(const float4*)(g + tid * 4);
  size_t o = (size_t)row * H_ + tid * 4;
  out[o + 0] = (__bf16)(f.x * gv.x * inv);
  out[o + 1] = (__bf16)(f.y * gv.y * inv);
  out[o + 2] = (__bf16)(f.z * gv.z * inv);
  out[o + 3] = (__bf16)(f.w * gv.w * inv);
}

// ---------------------------------------------------------------------------
// bf16 WMMA GEMM: C[M,N] = A[M,K] @ Bt[N,K]^T  (both operands K-contiguous).
// 128x128 block tile, BK=32, double-buffered LDS.
//   A tile staged with GLOBAL_LOAD_ASYNC_TO_LDS_B128 (ASYNCcnt),
//   B tile staged with the Tensor Data Mover (TENSORcnt) when available.
// 256 threads = 8 waves; wave owns 32x64 = 2x4 WMMA accumulators.
// ---------------------------------------------------------------------------
__global__ __launch_bounds__(256)
void gemm_bf16_wmma_kernel(const __bf16* __restrict__ A,
                           const __bf16* __restrict__ Bt,
                           int M, int N, int K,
                           float* __restrict__ outF,
                           __bf16* __restrict__ outB,
                           const float* __restrict__ resid) {
  __shared__ __bf16 sA[2][128][32];   // [buf][m][k], row = 64B
  __shared__ __bf16 sB[2][128][32];   // [buf][n][k], row = 64B
  const int tid  = threadIdx.x;
  const int lane = tid & 31;
  const int wave = tid >> 5;
  const int wr   = wave >> 1;           // 0..3
  const int wc   = wave & 1;            // 0..1
  const int bm   = blockIdx.y * 128;
  const int bn   = blockIdx.x * 128;
  const int lr   = lane & 15;
  const int kbA  = (lane & 16) ? 8 : 0;   // ISA 16-bit A-layout k-chunk base
  const int kbB  = (lane & 16) ? 16 : 0;  // contiguous-order k base (k = 16h + s), used for A and B

  const __bf16* Ab = A  + (size_t)bm * K;
  const __bf16* Bb = Bt + (size_t)bn * K;

  // Stage one 128x32 bf16 stripe (identical shape for A and B tiles).
  auto stage_stripe = [&](__bf16 (*s)[32], const __bf16* gbase, int k0) {
    #pragma unroll
    for (int i = 0; i < 2; ++i) {
      int g = tid + i * 256;                 // 512 chunks of 16B
      int row = g >> 2, kc = (g & 3) * 8;
      const __bf16* src = gbase + (size_t)row * K + k0 + kc;
      __bf16* dst = &s[row][kc];
#if defined(HAVE_ASYNC_LDS)
      __builtin_amdgcn_global_load_async_to_lds_b128(
          (AS1 v4i*)(uintptr_t)src,
          (AS3 v4i*)(uint32_t)(uintptr_t)dst, 0, 0);
#else
      *(v4u*)dst = *(const v4u*)src;
#endif
    }
  };

#if defined(HAVE_TDM)
  // TDM: move the whole 128x32 B stripe with one tensor_load_to_lds.
  auto stage_b_tdm = [&](int buf, int k0) {
    if (wave == 0) {
      uint64_t ga = (uint64_t)(uintptr_t)(Bb + k0);
      uint32_t la = (uint32_t)(uintptr_t)(&sB[buf][0][0]);
      v4u g0 = { 1u,                                   // count=1 (valid D#)
                 la,                                   // lds_addr
                 (unsigned)(ga & 0xFFFFFFFFu),         // global_addr[31:0]
                 ((unsigned)(ga >> 32) & 0x01FFFFFFu)  // global_addr[56:32]
                     | 0x80000000u };                  // type=2 ("image")
      v8i g1 = { (int)0x00010000,                      // data_size=1 -> 2 bytes
                 (int)((K & 0xFFFF) << 16),            // tensor_dim0[15:0]
                 (int)(((unsigned)K >> 16) | ((N & 0xFFFF) << 16)), // dim0 hi | dim1 lo
                 (int)(((unsigned)N >> 16) | (32u << 16)),          // dim1 hi | tile_dim0=32
                 (int)128,                             // tile_dim1=128, tile_dim2=0
                 (int)K,                               // tensor_dim0_stride[31:0]
                 0, 0 };                               // stride hi / dim1_stride
      v4i z4 = { 0, 0, 0, 0 };
#if __clang_major__ >= 23
      v8i z8 = { 0, 0, 0, 0, 0, 0, 0, 0 };
      __builtin_amdgcn_tensor_load_to_lds(g0, g1, z4, z4, z8, 0);
#else
      __builtin_amdgcn_tensor_load_to_lds(g0, g1, z4, z4, 0);
#endif
    }
  };
#endif

  auto stage = [&](int buf, int k0) {
    stage_stripe(sA[buf], Ab, k0);
#if defined(HAVE_TDM)
    stage_b_tdm(buf, k0);
#else
    stage_stripe(sB[buf], Bb, k0);
#endif
  };

  const v8f vz = {0.f, 0.f, 0.f, 0.f, 0.f, 0.f, 0.f, 0.f};
  v8f acc[2][4];
  #pragma unroll
  for (int i = 0; i < 2; ++i)
    #pragma unroll
    for (int j = 0; j < 4; ++j) acc[i][j] = vz;

  const int nk = K >> 5;
  stage(0, 0);

  for (int kt = 0; kt < nk; ++kt) {
    const int cur = kt & 1;
    wait_staging();
    __syncthreads();
    if (kt + 1 < nk) {
      stage(cur ^ 1, (kt + 1) * 32);
      __builtin_prefetch(Ab + (size_t)(tid >> 1) * K + (kt + 2) * 32, 0, 1);
    }

    FragBF af[2];
    #pragma unroll
    for (int mi = 0; mi < 2; ++mi) {
      const __bf16* ap = &sA[cur][wr * 32 + mi * 16 + lr][0];
      af[mi].u[0] = *(const v4u*)(ap + kbB);
      af[mi].u[1] = *(const v4u*)(ap + kbB + 8);
    }
    #pragma unroll
    for (int ni = 0; ni < 4; ++ni) {
      const __bf16* bp = &sB[cur][wc * 64 + ni * 16 + lr][kbB];
      FragBF bf;
      bf.u[0] = *(const v4u*)(bp);
      bf.u[1] = *(const v4u*)(bp + 8);
      #pragma unroll
      for (int mi = 0; mi < 2; ++mi)
        acc[mi][ni] = __builtin_amdgcn_wmma_f32_16x16x32_bf16(
            false, af[mi].v, false, bf.v, (short)0, acc[mi][ni], false, false);
    }
  }

  const int hi8 = (lane >> 4) * 8;
  #pragma unroll
  for (int mi = 0; mi < 2; ++mi)
    #pragma unroll
    for (int ni = 0; ni < 4; ++ni)
      #pragma unroll
      for (int r = 0; r < 8; ++r) {
        int m = bm + wr * 32 + mi * 16 + r + hi8;
        int n = bn + wc * 64 + ni * 16 + lr;
        float v = acc[mi][ni][r];
        if (outB) {
          outB[(size_t)m * N + n] = (__bf16)v;
        } else {
          float rv = resid ? resid[(size_t)m * N + n] : 0.f;
          outF[(size_t)m * N + n] = v + rv;
        }
      }
}

// ---------------------------------------------------------------------------
// RoPE + split qkv[M,3H] (bf16) -> q,k head-major [B,NH,S,HD]; V transposed
// [B,NH,HD,S] so flash-attention V-fragments are contiguous in the key dim.
// ---------------------------------------------------------------------------
__global__ void rope_split_kernel(const __bf16* __restrict__ qkv,
                                  __bf16* __restrict__ q,
                                  __bf16* __restrict__ k,
                                  __bf16* __restrict__ vt) {
  int idx = blockIdx.x * blockDim.x + threadIdx.x;  // M_*H_
  int row = idx >> 10, col = idx & 1023;
  int h = col >> 6, d = col & 63;
  int b = row >> 11, s = row & (S_ - 1);
  const __bf16* base = qkv + (size_t)row * (3 * H_);
  float qv = (float)base[col];
  float kv = (float)base[H_ + col];
  float vv = (float)base[2 * H_ + col];
  float qp = (float)base[col ^ 32];          // rotate-half partner (d ^ 32)
  float kp = (float)base[H_ + (col ^ 32)];
  int j = d & 31;
  float freq = __expf(-(float)(2 * j) * (9.2103403719761836f / 64.f));
  float ang = (float)s * freq;
  float sn, cs;
  __sincosf(ang, &sn, &cs);
  float sign = (d < 32) ? -1.f : 1.f;
  int bh = b * NH_ + h;
  size_t o = ((size_t)bh * S_ + s) * HD_ + d;
  q[o] = (__bf16)(qv * cs + sign * qp * sn);
  k[o] = (__bf16)(kv * cs + sign * kp * sn);
  vt[((size_t)bh * HD_ + d) * S_ + s] = (__bf16)vv;   // [bh][d][s]
}

// ---------------------------------------------------------------------------
// Flash attention (causal). Block = (b,h) x 64 q rows, 4 waves x 16 rows.
// Q/K fragments straight from global ([s][d], 32B/lane); V fragments straight
// from pre-transposed global ([d][s], 32B/lane). Only P goes through LDS.
// ---------------------------------------------------------------------------
__global__ __launch_bounds__(128)
void flash_attn_kernel(const __bf16* __restrict__ Q,
                       const __bf16* __restrict__ Kc,
                       const __bf16* __restrict__ Vt,
                       __bf16* __restrict__ O) {
  __shared__ __bf16 sP[4][16][32];    // per-wave P tile (A-layout staging)
  const int tid  = threadIdx.x;
  const int lane = tid & 31;
  const int wave = tid >> 5;
  const int bh   = blockIdx.y;
  const int b    = bh >> 4;
  const int h    = bh & 15;
  const int qBase = blockIdx.x * 64;
  const __bf16* Qp  = Q  + (size_t)bh * S_ * HD_;
  const __bf16* Kp  = Kc + (size_t)bh * S_ * HD_;
  const __bf16* VtP = Vt + (size_t)bh * HD_ * S_;
  const int lr  = lane & 15;
  const int kbA = (lane & 16) ? 8 : 0;
  const int kbB = (lane & 16) ? 16 : 0;
  const int hi8 = (lane >> 4) * 8;
  const int m0  = qBase + wave * 16;

  // Q fragments (16x64) straight from global
  FragBF aQ[2];
  {
    const __bf16* qrow = Qp + (size_t)(m0 + lr) * HD_;
    aQ[0].u[0] = *(const v4u*)(qrow + kbB);
    aQ[0].u[1] = *(const v4u*)(qrow + kbB + 8);
    aQ[1].u[0] = *(const v4u*)(qrow + 32 + kbB);
    aQ[1].u[1] = *(const v4u*)(qrow + 40 + kbB);
  }

  const v8f vz = {0.f, 0.f, 0.f, 0.f, 0.f, 0.f, 0.f, 0.f};
  v8f accO[4] = {vz, vz, vz, vz};
  float rmax[8], rsum[8];
  #pragma unroll
  for (int r = 0; r < 8; ++r) { rmax[r] = -3.0e38f; rsum[r] = 0.f; }

  for (int kc = 0; kc < qBase + 64; kc += 32) {
    // speculative prefetch of the next K chunk
    __builtin_prefetch(Kp + (size_t)(kc + 32 + (tid >> 2)) * HD_, 0, 1);

    float sc0[8], sc1[8];
    #pragma unroll
    for (int sub = 0; sub < 2; ++sub) {
      v8f sc = vz;
      int key = kc + sub * 16 + lr;
      const __bf16* kp = Kp + (size_t)key * HD_;
      FragBF bk0, bk1;
      bk0.u[0] = *(const v4u*)(kp + kbB);
      bk0.u[1] = *(const v4u*)(kp + kbB + 8);
      bk1.u[0] = *(const v4u*)(kp + 32 + kbB);
      bk1.u[1] = *(const v4u*)(kp + 32 + kbB + 8);
      sc = __builtin_amdgcn_wmma_f32_16x16x32_bf16(
          false, aQ[0].v, false, bk0.v, (short)0, sc, false, false);
      sc = __builtin_amdgcn_wmma_f32_16x16x32_bf16(
          false, aQ[1].v, false, bk1.v, (short)0, sc, false, false);
      float* dst = sub ? sc1 : sc0;
      #pragma unroll
      for (int r = 0; r < 8; ++r) {
        int qrow = m0 + r + hi8;
        float s = sc[r] * 0.125f;            // HD^-0.5
        dst[r] = (key <= qrow) ? s : -3.0e38f;
      }
    }

    // Online softmax (16-lane reductions within each wave32 half)
    #pragma unroll
    for (int r = 0; r < 8; ++r) {
      float cm = fmaxf(sc0[r], sc1[r]);
      #pragma unroll
      for (int m = 1; m < 16; m <<= 1) cm = fmaxf(cm, __shfl_xor(cm, m, 32));
      float nm   = fmaxf(rmax[r], cm);
      float corr = __expf(rmax[r] - nm);
      float p0 = __expf(sc0[r] - nm);
      float p1 = __expf(sc1[r] - nm);
      float ps = p0 + p1;
      #pragma unroll
      for (int m = 1; m < 16; m <<= 1) ps += __shfl_xor(ps, m, 32);
      rsum[r] = rsum[r] * corr + ps;
      rmax[r] = nm;
      #pragma unroll
      for (int t = 0; t < 4; ++t) accO[t][r] = accO[t][r] * corr;
      sP[wave][r + hi8][lr]      = (__bf16)p0;
      sP[wave][r + hi8][16 + lr] = (__bf16)p1;
    }
    __syncthreads();

    FragBF aP;
    const __bf16* pp = &sP[wave][lr][0];
    aP.u[0] = *(const v4u*)(pp + kbB);
    aP.u[1] = *(const v4u*)(pp + kbB + 8);
    #pragma unroll
    for (int t = 0; t < 4; ++t) {
      // V fragment: column d = t*16+lr, 16 contiguous keys from [d][s] layout
      const __bf16* vp = VtP + (size_t)(t * 16 + lr) * S_ + kc + kbB;
      FragBF bv;
      bv.u[0] = *(const v4u*)(vp);
      bv.u[1] = *(const v4u*)(vp + 8);
      accO[t] = __builtin_amdgcn_wmma_f32_16x16x32_bf16(
          false, aP.v, false, bv.v, (short)0, accO[t], false, false);
    }
    __syncthreads();
  }

  // Normalize and write bf16 attention output, row-major [B,S,H]
  #pragma unroll
  for (int t = 0; t < 4; ++t)
    #pragma unroll
    for (int r = 0; r < 8; ++r) {
      int srow = m0 + r + hi8;
      float val = accO[t][r] / rsum[r];
      size_t o = ((size_t)b * S_ + srow) * H_ + h * HD_ + t * 16 + lr;
      O[o] = (__bf16)val;
    }
}

// ---------------------------------------------------------------------------
// SwiGLU fuse: hh = a * silu(gate)
// ---------------------------------------------------------------------------
__global__ void swiglu_kernel(const __bf16* __restrict__ hgate,
                              __bf16* __restrict__ out) {
  int idx = blockIdx.x * blockDim.x + threadIdx.x;  // M_*FFN_
  int row = idx >> 12, j = idx & 4095;
  float a = (float)hgate[(size_t)row * (2 * FFN_) + j];
  float g = (float)hgate[(size_t)row * (2 * FFN_) + FFN_ + j];
  float s = g / (1.f + __expf(-g));
  out[(size_t)row * FFN_ + j] = (__bf16)(a * s);
}

// ---------------------------------------------------------------------------
// Launcher
// ---------------------------------------------------------------------------
extern "C" void kernel_launch(void* const* d_in, const int* in_sizes, int n_in,
                              void* d_out, int out_size, void* d_ws, size_t ws_size,
                              hipStream_t stream) {
  (void)in_sizes; (void)n_in; (void)out_size; (void)ws_size;
  const float* x    = (const float*)d_in[0];
  /* d_in[1] = mask (causal, implicit) */
  const float* g1   = (const float*)d_in[2];
  const float* g2   = (const float*)d_in[3];
  const float* Wqkv = (const float*)d_in[4];
  const float* Wout = (const float*)d_in[5];
  const float* W1   = (const float*)d_in[6];
  const float* W2   = (const float*)d_in[7];
  float* out = (float*)d_out;

  char* ws = (char*)d_ws;
  size_t off = 0;
  auto take = [&](size_t bytes) -> char* {
    char* p = ws + off;
    off += (bytes + 255) & ~(size_t)255;
    return p;
  };
  __bf16* wqkv_t = (__bf16*)take((size_t)H_ * 3 * H_ * 2);    // [3H][H]
  __bf16* wout_t = (__bf16*)take((size_t)H_ * H_ * 2);        // [H][H]
  __bf16* w1_t   = (__bf16*)take((size_t)H_ * 2 * FFN_ * 2);  // [2F][H]
  __bf16* w2_t   = (__bf16*)take((size_t)FFN_ * H_ * 2);      // [H][F]
  __bf16* xn_bf  = (__bf16*)take((size_t)M_ * H_ * 2);
  __bf16* qkv_bf = (__bf16*)take((size_t)M_ * 3 * H_ * 2);
  __bf16* q_bf   = (__bf16*)take((size_t)M_ * H_ * 2);
  __bf16* k_bf   = (__bf16*)take((size_t)M_ * H_ * 2);
  __bf16* vt_bf  = (__bf16*)take((size_t)M_ * H_ * 2);
  __bf16* ao_bf  = (__bf16*)take((size_t)M_ * H_ * 2);
  float*  x1     = (float*) take((size_t)M_ * H_ * 4);
  __bf16* xn2_bf = (__bf16*)take((size_t)M_ * H_ * 2);
  __bf16* h_bf   = (__bf16*)take((size_t)M_ * 2 * FFN_ * 2);
  __bf16* hh_bf  = (__bf16*)take((size_t)M_ * FFN_ * 2);

  // 0) Cast + transpose weights to bf16 [N][K]
  {
    int n;
    n = H_ * 3 * H_;
    cast_transpose_bf16_kernel<<<n / 256, 256, 0, stream>>>(Wqkv, wqkv_t, H_, 3 * H_);
    n = H_ * H_;
    cast_transpose_bf16_kernel<<<n / 256, 256, 0, stream>>>(Wout, wout_t, H_, H_);
    n = H_ * 2 * FFN_;
    cast_transpose_bf16_kernel<<<n / 256, 256, 0, stream>>>(W1, w1_t, H_, 2 * FFN_);
    n = FFN_ * H_;
    cast_transpose_bf16_kernel<<<n / 256, 256, 0, stream>>>(W2, w2_t, FFN_, H_);
  }

  // 1) xn = rmsnorm(x, g1)
  rmsnorm_bf16_kernel<<<M_, 256, 0, stream>>>(x, g1, xn_bf);

  // 2) qkv = xn @ Wqkv
  {
    dim3 g(3 * H_ / 128, M_ / 128);
    gemm_bf16_wmma_kernel<<<g, 256, 0, stream>>>(xn_bf, wqkv_t, M_, 3 * H_, H_,
                                                 nullptr, qkv_bf, nullptr);
  }

  // 3) RoPE + head-major split (V transposed)
  rope_split_kernel<<<(M_ * H_) / 256, 256, 0, stream>>>(qkv_bf, q_bf, k_bf, vt_bf);

  // 4) Causal flash attention
  {
    dim3 g(S_ / 64, B_ * NH_);
    flash_attn_kernel<<<g, 128, 0, stream>>>(q_bf, k_bf, vt_bf, ao_bf);
  }

  // 5) x1 = x + ao @ Wout
  {
    dim3 g(H_ / 128, M_ / 128);
    gemm_bf16_wmma_kernel<<<g, 256, 0, stream>>>(ao_bf, wout_t, M_, H_, H_,
                                                 x1, nullptr, x);
  }

  // 6) xn2 = rmsnorm(x1, g2)
  rmsnorm_bf16_kernel<<<M_, 256, 0, stream>>>(x1, g2, xn2_bf);

  // 7) h = xn2 @ W1
  {
    dim3 g(2 * FFN_ / 128, M_ / 128);
    gemm_bf16_wmma_kernel<<<g, 256, 0, stream>>>(xn2_bf, w1_t, M_, 2 * FFN_, H_,
                                                 nullptr, h_bf, nullptr);
  }

  // 8) hh = a * silu(gate)
  swiglu_kernel<<<(M_ * FFN_) / 256, 256, 0, stream>>>(h_bf, hh_bf);

  // 9) out = x1 + hh @ W2
  {
    dim3 g(H_ / 128, M_ / 128);
    gemm_bf16_wmma_kernel<<<g, 256, 0, stream>>>(hh_bf, w2_t, M_, H_, FFN_,
                                                 out, nullptr, x1);
  }
}
